// Model_79190607004152
// MI455X (gfx1250) — matrix operand order repaired and checked by reading, not compiled, not run on hardware
//
#include <hip/hip_runtime.h>
#include <hip/hip_bf16.h>

typedef __attribute__((ext_vector_type(16))) __bf16 v16bf;
typedef __attribute__((ext_vector_type(8)))  __bf16 v8bf;
typedef __attribute__((ext_vector_type(8)))  float  v8f;

union V16 { v16bf v; v8bf h[2]; };

#define LL     2048
#define DKK    64
#define HHN    8
#define GGN    4
#define LOCALH 4
#define DDIM   512
#define NREL   (2*LL - 1)
#define NEGBIG (-1.0e12f)
#define LOG2E  1.4426950408889634f
#define NELEM  (4 * HHN * LL * DKK)   // 4,194,304 elements per tensor

// ---------------- Precompute kernels (outputs pre-scaled by log2e) ----------------

__global__ void rel_k(const float* __restrict__ relw, const float* __restrict__ mlpw,
                      float* __restrict__ out) {
    int idx = blockIdx.x * blockDim.x + threadIdx.x;
    if (idx >= NREL * LOCALH) return;
    int r = idx >> 2, hg = idx & 3;
    float acc = 0.f;
    #pragma unroll 8
    for (int d = 0; d < DKK; ++d)
        acc += relw[(size_t)r * (LOCALH * DKK) + hg * DKK + d] * mlpw[hg * DKK + d];
    out[r * LOCALH + hg] = acc * LOG2E;
}

__global__ void vterm_k(const float* __restrict__ V, const float* __restrict__ mlpw,
                        float* __restrict__ out) {
    int idx = blockIdx.x * blockDim.x + threadIdx.x;
    if (idx >= 4 * LOCALH * LL) return;
    int k = idx & (LL - 1), hl = (idx >> 11) & 3, b = idx >> 13;
    const float* vp = V + (((size_t)(b * HHN + GGN + hl)) * LL + k) * DKK;
    float acc = 0.f;
    #pragma unroll 8
    for (int d = 0; d < DKK; ++d) acc += vp[d] * mlpw[hl * DKK + d];
    out[idx] = acc * LOG2E;
}

__global__ void wvec_k(const float* __restrict__ upw, const float* __restrict__ mlpw,
                       float* __restrict__ out) {
    int idx = blockIdx.x * blockDim.x + threadIdx.x;
    if (idx >= LOCALH * DDIM) return;
    int hl = idx >> 9, d2 = idx & (DDIM - 1);
    float acc = 0.f;
    #pragma unroll 8
    for (int d = 0; d < DKK; ++d)
        acc += mlpw[hl * DKK + d] * upw[(size_t)(hl * DKK + d) * DDIM + d2];
    out[idx] = acc;
}

__global__ void qadd_k(const float* __restrict__ users, const float* __restrict__ upb,
                       const float* __restrict__ mlpw, const float* __restrict__ mlpb,
                       const float* __restrict__ wvec, const float* __restrict__ vterm,
                       float* __restrict__ out) {
    int idx = blockIdx.x * blockDim.x + threadIdx.x;
    if (idx >= 4 * LOCALH * LL) return;
    int q = idx & (LL - 1), hl = (idx >> 11) & 3, b = idx >> 13;
    const float* up  = users + ((size_t)b * LL + q) * DDIM;
    const float* wvp = wvec + hl * DDIM;
    float acc = 0.f;
    #pragma unroll 8
    for (int i = 0; i < DDIM; ++i) acc += up[i] * wvp[i];
    float bc = 0.f;
    #pragma unroll 8
    for (int d = 0; d < DKK; ++d) bc += mlpw[hl * DKK + d] * upb[hl * DKK + d];
    out[idx] = vterm[idx] + LOG2E * (acc + bc + mlpb[hl]);
}

__global__ __launch_bounds__(256) void qk_cvt(const float* __restrict__ Q,
                                              const float* __restrict__ K,
                                              __bf16* __restrict__ Qb,
                                              __bf16* __restrict__ Kb) {
    const float qs = 0.125f * LOG2E;
    int e = (blockIdx.x * blockDim.x + threadIdx.x) * 4;
    float4 q = *(const float4*)(Q + e);
    float4 k = *(const float4*)(K + e);
    Qb[e]     = (__bf16)(q.x * qs);
    Qb[e + 1] = (__bf16)(q.y * qs);
    Qb[e + 2] = (__bf16)(q.z * qs);
    Qb[e + 3] = (__bf16)(q.w * qs);
    Kb[e]     = (__bf16)k.x;
    Kb[e + 1] = (__bf16)k.y;
    Kb[e + 2] = (__bf16)k.z;
    Kb[e + 3] = (__bf16)k.w;
}

__global__ __launch_bounds__(256) void vt_k(const float* __restrict__ V,
                                            __bf16* __restrict__ Vt) {
    __shared__ __bf16 tile[64][65];
    const int bh = blockIdx.x >> 5;
    const int kt = blockIdx.x & 31;
    const int tid = threadIdx.x;
    const int r0 = tid >> 2;
    const int c0 = (tid & 3) * 16;
    const float* src = V + ((size_t)bh * LL + kt * 64 + r0) * DKK + c0;
    #pragma unroll
    for (int i = 0; i < 16; ++i) tile[r0][c0 + i] = (__bf16)src[i];
    __syncthreads();
    __bf16* dst = Vt + ((size_t)bh * DKK + r0) * LL + kt * 64 + c0;
    #pragma unroll
    for (int i = 0; i < 16; ++i) dst[i] = tile[c0 + i][r0];
}

// ---------------- Flash-attention main kernel (bf16 WMMA, exp2 domain) ----------------

__global__ __launch_bounds__(256) void attn_main(
    const __bf16* __restrict__ Qb, const __bf16* __restrict__ Kb,
    const __bf16* __restrict__ Vt, const float* __restrict__ rel_scores,
    const float* __restrict__ vterm, const float* __restrict__ qadd,
    float* __restrict__ out) {
    __shared__ float rel_s[NREL];
    __shared__ alignas(16) __bf16 pst[8][16 * 64];

    const int tid  = threadIdx.x;
    const int lane = tid & 31;
    const int wv   = tid >> 5;
    const int l16  = lane & 15;
    const int hh   = lane >> 4;

    const int bh = blockIdx.x >> 4;
    const int tg = blockIdx.x & 15;
    const int b  = bh >> 3;
    const int h  = bh & 7;
    const bool is_local = (h >= GGN);
    const int hl = h - GGN;

    if (is_local) {
        for (int i = tid; i < NREL; i += 256) rel_s[i] = rel_scores[i * LOCALH + hl];
    }
    __syncthreads();

    const size_t base = (size_t)bh * LL * DKK;
    const __bf16* Qp = Qb + base;
    const __bf16* Kp = Kb + base;
    const __bf16* Vp = Vt + base;   // [d][k]
    const float*  vtb = vterm + (size_t)(b * LOCALH + hl) * LL;

    const int qbase = tg * 128 + wv * 16;

    V16 qa[2];
    {
        const int qrow = qbase + l16;
        #pragma unroll
        for (int c = 0; c < 2; ++c) {
            const __bf16* qp = Qp + (size_t)qrow * DKK + c * 32 + hh * 16;
            qa[c].h[0] = *(const v8bf*)qp;
            qa[c].h[1] = *(const v8bf*)(qp + 8);
        }
    }

    float rowc[8];
    #pragma unroll
    for (int r = 0; r < 8; ++r)
        rowc[r] = is_local ? qadd[(size_t)(b * LOCALH + hl) * LL + qbase + hh * 8 + r] : 0.0f;

    v16bf onesv;
    #pragma unroll
    for (int i = 0; i < 16; ++i) onesv[i] = (__bf16)1.0f;

    v8f o[4], o4;
    const v8f vzero = {0.f, 0.f, 0.f, 0.f, 0.f, 0.f, 0.f, 0.f};
    #pragma unroll
    for (int d = 0; d < 4; ++d) o[d] = vzero;
    o4 = vzero;

    // ---- one 64-key block; inlined twice (steady-state / diagonal) ----
    auto do_block = [&](const int kbase, const bool domask) {
        // K fragments: batched so the S phase takes one clustered wait
        v16bf kf[4][2];
        #pragma unroll
        for (int t = 0; t < 4; ++t) {
            const __bf16* kr = Kp + (size_t)(kbase + t * 16 + l16) * DKK + hh * 16;
            kf[t][0] = *(const v16bf*)kr;
            kf[t][1] = *(const v16bf*)(kr + 32);
        }
        v8f s[4];
        #pragma unroll
        for (int t = 0; t < 4; ++t) {
            s[t] = __builtin_amdgcn_wmma_f32_16x16x32_bf16(false, qa[0].v, false, kf[t][0],
                                                           (short)0, vzero, false, false);
            s[t] = __builtin_amdgcn_wmma_f32_16x16x32_bf16(false, qa[1].v, false, kf[t][1],
                                                           (short)0, s[t], false, false);
        }

        // V fragments issued NOW: latency hides under the whole softmax phase
        v16bf vf[2][4];
        #pragma unroll
        for (int kh = 0; kh < 2; ++kh)
            #pragma unroll
            for (int d = 0; d < 4; ++d)
                vf[kh][d] = *(const v16bf*)(Vp + (size_t)(d * 16 + l16) * LL +
                                            kbase + kh * 32 + hh * 16);
        __builtin_amdgcn_sched_barrier(0);   // keep V loads issued here

        if (is_local) {
            #pragma unroll
            for (int t = 0; t < 4; ++t) {
                const int kcol = kbase + t * 16 + l16;
                const float vk = vtb[kcol];
                #pragma unroll
                for (int r = 0; r < 8; ++r) {
                    const int qr = qbase + hh * 8 + r;
                    s[t][r] += rel_s[kcol - qr + (LL - 1)] + vk + rowc[r];
                }
            }
        }
        if (domask) {
            #pragma unroll
            for (int t = 0; t < 4; ++t) {
                const int kcol = kbase + t * 16 + l16;
                #pragma unroll
                for (int r = 0; r < 8; ++r)
                    if (kcol > qbase + hh * 8 + r) s[t][r] = NEGBIG;
            }
        }

        // P = exp2(S) -> bf16, transpose through LDS
        #pragma unroll
        for (int t = 0; t < 4; ++t)
            #pragma unroll
            for (int r = 0; r < 8; ++r)
                pst[wv][(hh * 8 + r) * 64 + t * 16 + l16] = (__bf16)exp2f(s[t][r]);
        asm volatile("s_wait_dscnt 0" ::: "memory");

        V16 pf[2];
        #pragma unroll
        for (int kh = 0; kh < 2; ++kh) {
            const __bf16* pp = &pst[wv][l16 * 64 + kh * 32 + hh * 16];
            pf[kh].h[0] = *(const v8bf*)pp;
            pf[kh].h[1] = *(const v8bf*)(pp + 8);
        }
        asm volatile("" ::: "memory");

        // O += P*V, denominator += P*1 — V already resident
        #pragma unroll
        for (int kh = 0; kh < 2; ++kh) {
            #pragma unroll
            for (int d = 0; d < 4; ++d)
                o[d] = __builtin_amdgcn_wmma_f32_16x16x32_bf16(false, pf[kh].v, false,
                                                               vf[kh][d], (short)0, o[d],
                                                               false, false);
            o4 = __builtin_amdgcn_wmma_f32_16x16x32_bf16(false, pf[kh].v, false, onesv,
                                                         (short)0, o4, false, false);
        }
    };

    const int nblk = (qbase + 16 + 63) >> 6;
    for (int j = 0; j < nblk - 1; ++j) do_block(j << 6, false);
    do_block((nblk - 1) << 6, true);   // diagonal block: causal mask

    float* Ob = out + base;
    #pragma unroll
    for (int r = 0; r < 8; ++r) {
        const float inv = 1.0f / o4[r];
        const int qr = qbase + hh * 8 + r;
        #pragma unroll
        for (int d = 0; d < 4; ++d)
            Ob[(size_t)qr * DKK + d * 16 + l16] = o[d][r] * inv;
    }
}

// ---------------- Launch ----------------

extern "C" void kernel_launch(void* const* d_in, const int* in_sizes, int n_in,
                              void* d_out, int out_size, void* d_ws, size_t ws_size,
                              hipStream_t stream) {
    const float* Q     = (const float*)d_in[0];
    const float* K     = (const float*)d_in[1];
    const float* V     = (const float*)d_in[2];
    const float* users = (const float*)d_in[3];
    // d_in[4] = causal mask (computed analytically, not read)
    const float* relw  = (const float*)d_in[5];
    const float* upw   = (const float*)d_in[6];
    const float* upb   = (const float*)d_in[7];
    const float* mlpw  = (const float*)d_in[8];
    const float* mlpb  = (const float*)d_in[9];

    float* ws         = (float*)d_ws;
    float* rel_scores = ws;
    float* vterm      = rel_scores + NREL * LOCALH;
    float* qadd       = vterm + 4 * LOCALH * LL;
    float* wvec       = qadd + 4 * LOCALH * LL;
    size_t foff = ((size_t)(NREL * LOCALH + 8 * LOCALH * LL + LOCALH * DDIM) * 4 + 255) & ~(size_t)255;
    __bf16* Qb16 = (__bf16*)((char*)d_ws + foff);
    __bf16* Kb16 = Qb16 + NELEM;
    __bf16* Vt   = Kb16 + NELEM;

    qk_cvt<<<NELEM / 4 / 256, 256, 0, stream>>>(Q, K, Qb16, Kb16);
    vt_k  <<<4 * HHN * (LL / 64), 256, 0, stream>>>(V, Vt);

    rel_k  <<<(NREL * LOCALH + 255) / 256, 256, 0, stream>>>(relw, mlpw, rel_scores);
    vterm_k<<<(4 * LOCALH * LL + 255) / 256, 256, 0, stream>>>(V, mlpw, vterm);
    wvec_k <<<(LOCALH * DDIM + 255) / 256, 256, 0, stream>>>(upw, mlpw, wvec);
    qadd_k <<<(4 * LOCALH * LL + 255) / 256, 256, 0, stream>>>(users, upb, mlpw, mlpb,
                                                               wvec, vterm, qadd);

    attn_main<<<4 * HHN * (LL / 128), 256, 0, stream>>>(Qb16, Kb16, Vt, rel_scores,
                                                        vterm, qadd, (float*)d_out);
}
